// MaxwellDemonFilter_87402584473613
// MI455X (gfx1250) — compile-verified, matrix operand order checked by reading, not run on hardware
//
#include <hip/hip_runtime.h>
#include <hip/hip_bf16.h>
#include <math.h>

#define NN 50000
#define EE 800000
#define NEG_SLOPE 0.2f

typedef __attribute__((ext_vector_type(16))) _Float16 v16h;
typedef __attribute__((ext_vector_type(8)))  float    v8f;

// ---------------------------------------------------------------------------
// Fragment loader for v_wmma_f32_16x16x32_f16 (wave32).
// A-matrix: lane&15 = row M, (lane>>4)*8 = K base; VGPR v holds
// K = (v>>2)*16 + kbase + (v&3)*2 (+1 for high half).
// B-matrix (from row-major W[Nout][K], since out = A @ W^T): identical
// pattern with lane&15 = output column n, rows of W.
// ---------------------------------------------------------------------------
__device__ __forceinline__ v16h load_frag16(const _Float16* base, int ld) {
  const int lane = threadIdx.x & 31;
  const int r    = lane & 15;
  const int kb   = (lane >> 4) << 3;
  const _Float16* row = base + r * ld + kb;
  v16h f;
#pragma unroll
  for (int v = 0; v < 8; ++v) {
    const int k = ((v >> 2) << 4) + ((v & 3) << 1);
    f[2 * v]     = row[k];
    f[2 * v + 1] = row[k + 1];
  }
  return f;
}

// Monotone float->uint encoding so unsigned atomicMax == float max and the
// all-zero (memset) state sits below every encodable real number.
__device__ __forceinline__ unsigned enc_f(float f) {
  unsigned u = __float_as_uint(f);
  return (u & 0x80000000u) ? ~u : (u | 0x80000000u);
}
__device__ __forceinline__ float dec_f(unsigned u) {
  return (u & 0x80000000u) ? __uint_as_float(u & 0x7fffffffu) : __uint_as_float(~u);
}

// ---------------------------------------------------------------------------
// Weight f32 -> f16 conversion (once per launch; keeps WMMA operand traffic
// at 2 B/elem and L2-resident).
// ---------------------------------------------------------------------------
__global__ void k_convert_weights(const float* __restrict__ fc_w,
                                  const float* __restrict__ l1_w,
                                  const float* __restrict__ l2_w,
                                  _Float16* __restrict__ wfc,
                                  _Float16* __restrict__ wl1,
                                  _Float16* __restrict__ wl2) {
  const int i = blockIdx.x * blockDim.x + threadIdx.x;
  if (i < 64 * 64)   wfc[i] = (_Float16)fc_w[i];
  if (i < 256 * 256) wl1[i] = (_Float16)l1_w[i];
  if (i < 64 * 256)  wl2[i] = (_Float16)l2_w[i];
}

// ---------------------------------------------------------------------------
// Node kernel 1: xfc = x @ fc_w^T + fc_b  (WMMA), then head projections
// as_u / as_v / laplace from the LDS-resident f32 tile.
// 128 threads = 4 waves; wave w owns output columns [16w, 16w+16).
// ---------------------------------------------------------------------------
__global__ void k_node_fc(const float* __restrict__ x,
                          const _Float16* __restrict__ wfc,
                          const float* __restrict__ fc_b,
                          const float* __restrict__ au_w, const float* __restrict__ au_b,
                          const float* __restrict__ av_w,
                          const float* __restrict__ gl_w, const float* __restrict__ gl_b,
                          const float* __restrict__ chaos_noise,
                          const float* __restrict__ chaos_factor,
                          float* __restrict__ xfc, float* __restrict__ asu,
                          float* __restrict__ asv, float* __restrict__ lap) {
  __shared__ _Float16 sh_x[16][64];
  __shared__ float    sh_fc[16][64];
  const int node0 = blockIdx.x * 16;
  const int tid   = threadIdx.x;

  for (int t = tid; t < 16 * 64; t += 128)
    sh_x[t >> 6][t & 63] = (_Float16)x[(node0 + (t >> 6)) * 64 + (t & 63)];
  __syncthreads();

  const int wave = tid >> 5;
  const int lane = tid & 31;
  v8f acc = {};
#pragma unroll
  for (int ks = 0; ks < 2; ++ks) {
    v16h a = load_frag16(&sh_x[0][ks * 32], 64);
    v16h b = load_frag16(wfc + wave * 16 * 64 + ks * 32, 64);
    acc = __builtin_amdgcn_wmma_f32_16x16x32_f16(false, a, false, b, (short)0, acc,
                                                 false, false);
  }
  {
    const int n   = lane & 15;
    const int col = wave * 16 + n;
    const int mB  = (lane >> 4) << 3;   // C/D: VGPR j -> M = j + 8*(lane>>4)
#pragma unroll
    for (int j = 0; j < 8; ++j) {
      const int m = mB + j;
      const float v = acc[j] + fc_b[col];
      xfc[(node0 + m) * 64 + col] = v;
      sh_fc[m][col] = v;
    }
  }
  __syncthreads();

  // 128 threads -> (row, head) pairs; 64-wide dots out of LDS.
  const int r = tid >> 3;
  const int h = tid & 7;
  float su = 0.f, sv = 0.f, sl = 0.f;
#pragma unroll 8
  for (int k = 0; k < 64; ++k) {
    const float xv = sh_fc[r][k];
    su += xv * au_w[h * 64 + k];
    sv += xv * av_w[h * 64 + k];
    sl += xv * gl_w[h * 64 + k];
  }
  const int node = node0 + r;
  asu[node * 8 + h] = su + au_b[h];
  asv[node * 8 + h] = sv;
  lap[node * 8 + h] = sl + gl_b[h] + chaos_noise[node * 8 + h] * chaos_factor[0];
}

// ---------------------------------------------------------------------------
// Edge pass 1: leaky score (stored), segment-max via encoded atomicMax,
// neighbor laplace sum, in/out degrees.
// ---------------------------------------------------------------------------
__global__ void k_edge_pass1(const int* __restrict__ src, const int* __restrict__ dst,
                             const float* __restrict__ asu, const float* __restrict__ asv,
                             const float* __restrict__ lap,
                             float* __restrict__ ebuf, unsigned* __restrict__ smax,
                             float* __restrict__ nbr,
                             float* __restrict__ indeg, float* __restrict__ outdeg) {
  const int e = blockIdx.x * blockDim.x + threadIdx.x;
  if (e >= EE) return;
  const int s = src[e], d = dst[e];
  atomicAdd(&indeg[d], 1.f);
  atomicAdd(&outdeg[s], 1.f);
#pragma unroll
  for (int h = 0; h < 8; ++h) {
    float sc = asu[s * 8 + h] + asv[d * 8 + h];
    sc = sc >= 0.f ? sc : NEG_SLOPE * sc;
    ebuf[e * 8 + h] = sc;
    atomicMax(&smax[d * 8 + h], enc_f(sc));
    atomicAdd(&nbr[d * 8 + h], lap[s * 8 + h]);
  }
}

__global__ void k_node_energy(const float* __restrict__ lap, const float* __restrict__ nbr,
                              const float* __restrict__ indeg, float* __restrict__ energy) {
  const int n = blockIdx.x * blockDim.x + threadIdx.x;
  if (n >= NN) return;
  const float id = fmaxf(indeg[n], 1.f);
#pragma unroll
  for (int h = 0; h < 8; ++h)
    energy[n * 8 + h] = lap[n * 8 + h] - nbr[n * 8 + h] / id;
}

// Edge pass 2: e = exp(score - smax[dst]); accumulate softmax denominator.
__global__ void k_edge_pass2(const int* __restrict__ dst, float* __restrict__ ebuf,
                             const unsigned* __restrict__ smax, float* __restrict__ ssum) {
  const int e = blockIdx.x * blockDim.x + threadIdx.x;
  if (e >= EE) return;
  const int d = dst[e];
#pragma unroll
  for (int h = 0; h < 8; ++h) {
    const float v = expf(ebuf[e * 8 + h] - dec_f(smax[d * 8 + h]));
    ebuf[e * 8 + h] = v;
    atomicAdd(&ssum[d * 8 + h], v);
  }
}

// ---------------------------------------------------------------------------
// Edge pass 3 (heavy scatter): one wave per edge; lane owns channels
// lane and lane+32 -> both map to head (lane & 7), so each lane needs one
// attention scalar. 3 weighted scatters per channel.
// ---------------------------------------------------------------------------
__global__ void k_edge_pass3(const int* __restrict__ src, const int* __restrict__ dst,
                             const float* __restrict__ ebuf, const float* __restrict__ ssum,
                             const float* __restrict__ energy, const float* __restrict__ outdeg,
                             const float* __restrict__ xfc,
                             float* __restrict__ msg1, float* __restrict__ msg2,
                             float* __restrict__ msg3) {
  const int e = blockIdx.x * 8 + (threadIdx.x >> 5);
  if (e >= EE) return;
  const int lane = threadIdx.x & 31;
  const int s = src[e], d = dst[e];
  const int h = lane & 7;
  // sigmoid(energy[d]-energy[s]) = 1/(1+exp(energy[s]-energy[d]))
  const float filt = 1.f / (1.f + expf(energy[s * 8 + h] - energy[d * 8 + h]));
  const float attn = (ebuf[e * 8 + h] / ssum[d * 8 + h]) * filt;
  const float prod = outdeg[s] * outdeg[d];
  const float nrm  = prod > 0.f ? rsqrtf(prod) : 0.f;
#pragma unroll
  for (int half = 0; half < 2; ++half) {
    const int c = lane + half * 32;       // reshape(N,HD,H): head = c % 8
    const float xv = xfc[s * 64 + c];
    atomicAdd(&msg1[d * 64 + c], xv * attn);
    atomicAdd(&msg2[d * 64 + c], xv);
    atomicAdd(&msg3[d * 64 + c], xv * nrm);
  }
}

// ---------------------------------------------------------------------------
// Final kernel: h = [xfc | msg1 | msg2/deg | msg3] (16x256 f16 tile in LDS),
// GEMM1 (256->256, WMMA, bias+exact GELU fused) -> LDS f16, GEMM2 (256->64).
// 256 threads = 8 waves: GEMM1 wave w does N-tiles {2w, 2w+1} sharing A frags;
// GEMM2 uses waves 0..3 (wave-uniform branch, EXEC all-ones inside).
// ---------------------------------------------------------------------------
__global__ void k_final(const float* __restrict__ xfc, const float* __restrict__ msg1,
                        const float* __restrict__ msg2, const float* __restrict__ msg3,
                        const float* __restrict__ indeg,
                        const _Float16* __restrict__ wl1, const float* __restrict__ l1_b,
                        const _Float16* __restrict__ wl2, const float* __restrict__ l2_b,
                        float* __restrict__ out) {
  __shared__ _Float16 sh_h[16][256];
  __shared__ _Float16 sh_mid[16][256];
  const int node0 = blockIdx.x * 16;
  const int tid   = threadIdx.x;

  const int c = tid, seg = c >> 6, cc = c & 63;
#pragma unroll 4
  for (int r = 0; r < 16; ++r) {
    const int node = node0 + r;
    float v;
    if (seg == 0)      v = xfc[node * 64 + cc];
    else if (seg == 1) v = msg1[node * 64 + cc];
    else if (seg == 2) v = msg2[node * 64 + cc] / fmaxf(indeg[node], 1.f);
    else               v = msg3[node * 64 + cc];
    sh_h[r][c] = (_Float16)v;
  }
  __syncthreads();

  const int wave = tid >> 5;
  const int lane = tid & 31;
  {
    v8f acc0 = {}, acc1 = {};
    const _Float16* b0 = wl1 + (size_t)(wave * 2 + 0) * 16 * 256;
    const _Float16* b1 = wl1 + (size_t)(wave * 2 + 1) * 16 * 256;
#pragma unroll
    for (int ks = 0; ks < 8; ++ks) {
      v16h a  = load_frag16(&sh_h[0][ks * 32], 256);
      v16h f0 = load_frag16(b0 + ks * 32, 256);
      v16h f1 = load_frag16(b1 + ks * 32, 256);
      acc0 = __builtin_amdgcn_wmma_f32_16x16x32_f16(false, a, false, f0, (short)0, acc0,
                                                    false, false);
      acc1 = __builtin_amdgcn_wmma_f32_16x16x32_f16(false, a, false, f1, (short)0, acc1,
                                                    false, false);
    }
    const int n  = lane & 15;
    const int mB = (lane >> 4) << 3;
#pragma unroll
    for (int j = 0; j < 8; ++j) {
      const int m = mB + j;
      {
        const int col = (wave * 2 + 0) * 16 + n;
        float g = acc0[j] + l1_b[col];
        g = 0.5f * g * (1.f + erff(g * 0.70710678118f));   // exact GELU
        sh_mid[m][col] = (_Float16)g;
      }
      {
        const int col = (wave * 2 + 1) * 16 + n;
        float g = acc1[j] + l1_b[col];
        g = 0.5f * g * (1.f + erff(g * 0.70710678118f));
        sh_mid[m][col] = (_Float16)g;
      }
    }
  }
  __syncthreads();

  if (wave < 4) {
    v8f acc = {};
    const _Float16* bb = wl2 + (size_t)wave * 16 * 256;
#pragma unroll
    for (int ks = 0; ks < 8; ++ks) {
      v16h a = load_frag16(&sh_mid[0][ks * 32], 256);
      v16h b = load_frag16(bb + ks * 32, 256);
      acc = __builtin_amdgcn_wmma_f32_16x16x32_f16(false, a, false, b, (short)0, acc,
                                                   false, false);
    }
    const int n   = lane & 15;
    const int col = wave * 16 + n;
    const int mB  = (lane >> 4) << 3;
#pragma unroll
    for (int j = 0; j < 8; ++j)
      out[(node0 + mB + j) * 64 + col] = acc[j] + l2_b[col];
  }
}

// ---------------------------------------------------------------------------
extern "C" void kernel_launch(void* const* d_in, const int* in_sizes, int n_in,
                              void* d_out, int out_size, void* d_ws, size_t ws_size,
                              hipStream_t stream) {
  (void)in_sizes; (void)n_in; (void)out_size; (void)ws_size;
  const float* x     = (const float*)d_in[0];
  const float* noise = (const float*)d_in[1];
  const float* fc_w  = (const float*)d_in[2];
  const float* fc_b  = (const float*)d_in[3];
  const float* au_w  = (const float*)d_in[4];
  const float* au_b  = (const float*)d_in[5];
  const float* av_w  = (const float*)d_in[6];
  const float* gl_w  = (const float*)d_in[7];
  const float* gl_b  = (const float*)d_in[8];
  const float* cfac  = (const float*)d_in[9];
  const float* l1_w  = (const float*)d_in[10];
  const float* l1_b  = (const float*)d_in[11];
  const float* l2_w  = (const float*)d_in[12];
  const float* l2_b  = (const float*)d_in[13];
  const int*   src   = (const int*)d_in[14];
  const int*   dst   = (const int*)d_in[15];
  float* out = (float*)d_out;

  // Workspace layout: zeroed accumulator region first (single memset, also
  // serves as -inf sentinel for the encoded segment-max), then scratch.
  char* p = (char*)d_ws;
  float* msg1 = (float*)p;        p += (size_t)NN * 64 * 4;
  float* msg2 = (float*)p;        p += (size_t)NN * 64 * 4;
  float* msg3 = (float*)p;        p += (size_t)NN * 64 * 4;
  float* nbr  = (float*)p;        p += (size_t)NN * 8 * 4;
  float* ssum = (float*)p;        p += (size_t)NN * 8 * 4;
  unsigned* smax = (unsigned*)p;  p += (size_t)NN * 8 * 4;
  float* indeg  = (float*)p;      p += (size_t)NN * 4;
  float* outdeg = (float*)p;      p += (size_t)NN * 4;
  const size_t zbytes = (size_t)(p - (char*)d_ws);
  float* xfc   = (float*)p;       p += (size_t)NN * 64 * 4;
  float* asu   = (float*)p;       p += (size_t)NN * 8 * 4;
  float* asv   = (float*)p;       p += (size_t)NN * 8 * 4;
  float* lap   = (float*)p;       p += (size_t)NN * 8 * 4;
  float* energy = (float*)p;      p += (size_t)NN * 8 * 4;
  float* ebuf  = (float*)p;       p += (size_t)EE * 8 * 4;
  _Float16* wfc = (_Float16*)p;   p += (size_t)64 * 64 * 2;
  _Float16* wl1 = (_Float16*)p;   p += (size_t)256 * 256 * 2;
  _Float16* wl2 = (_Float16*)p;   p += (size_t)64 * 256 * 2;

  hipMemsetAsync(d_ws, 0, zbytes, stream);

  k_convert_weights<<<(256 * 256 + 255) / 256, 256, 0, stream>>>(fc_w, l1_w, l2_w,
                                                                 wfc, wl1, wl2);
  k_node_fc<<<NN / 16, 128, 0, stream>>>(x, wfc, fc_b, au_w, au_b, av_w, gl_w, gl_b,
                                         noise, cfac, xfc, asu, asv, lap);
  k_edge_pass1<<<(EE + 255) / 256, 256, 0, stream>>>(src, dst, asu, asv, lap, ebuf,
                                                     smax, nbr, indeg, outdeg);
  k_node_energy<<<(NN + 255) / 256, 256, 0, stream>>>(lap, nbr, indeg, energy);
  k_edge_pass2<<<(EE + 255) / 256, 256, 0, stream>>>(dst, ebuf, smax, ssum);
  k_edge_pass3<<<(EE + 7) / 8, 256, 0, stream>>>(src, dst, ebuf, ssum, energy, outdeg,
                                                 xfc, msg1, msg2, msg3);
  k_final<<<NN / 16, 256, 0, stream>>>(xfc, msg1, msg2, msg3, indeg, wl1, l1_b,
                                       wl2, l2_b, out);
}
